// RecurrentNetTimeFixedOnlyRec_25503515803769
// MI455X (gfx1250) — matrix operand order repaired and checked by reading, not compiled, not run on hardware
//
#include <hip/hip_runtime.h>

#define B_   256
#define T_   500
#define NIN  128
#define NHID 512
#define NOUT 64

typedef __attribute__((ext_vector_type(16))) __bf16 v16bf;
typedef __attribute__((ext_vector_type(8)))  __bf16 v8bf;
typedef __attribute__((ext_vector_type(8)))  float  v8f;
typedef __attribute__((ext_vector_type(4)))  float  f4;
typedef __attribute__((vector_size(16)))     int    v4i_;

// workspace layout (bytes)
#define WS_CNT   0                         // unsigned barrier counter
#define WS_WOUT  256                       // NOUT*NHID bf16 = 65536 B
#define WS_HBUF  (256 + NOUT*NHID*2)       // 2 * B_*NHID bf16 = 524288 B

__device__ __forceinline__ v16bf cat8(v8bf lo, v8bf hi) {
  return __builtin_shufflevector(lo, hi, 0,1,2,3,4,5,6,7,8,9,10,11,12,13,14,15);
}
__device__ __forceinline__ v8f wmma_bf16(v16bf a, v16bf b, v8f c) {
  // 8 args: (neg_a, A, neg_b, B, c_mod, C, reuse_a, reuse_b)
  return __builtin_amdgcn_wmma_f32_16x16x32_bf16(false, a, false, b, (short)0, c, false, false);
}

// Async Global -> LDS copy of 16 bytes per lane (gfx1250 path, ASYNCcnt).
__device__ __forceinline__ void async_g2l_b128(const void* gptr, void* lptr) {
#if __has_builtin(__builtin_amdgcn_global_load_async_to_lds_b128)
  __builtin_amdgcn_global_load_async_to_lds_b128(
      (__attribute__((address_space(1))) v4i_*)(void*)gptr,
      (__attribute__((address_space(3))) v4i_*)lptr, 0, 0);
#else
  unsigned lo = (unsigned)(size_t)(__attribute__((address_space(3))) void*)lptr;
  asm volatile("global_load_async_to_lds_b128 %0, %1, off"
               :: "v"(lo), "v"(gptr) : "memory");
#endif
}
__device__ __forceinline__ void wait_async0() {
#if __has_builtin(__builtin_amdgcn_s_wait_asynccnt)
  __builtin_amdgcn_s_wait_asynccnt(0);
#else
  asm volatile("s_wait_asynccnt 0" ::: "memory");
#endif
}

__device__ __forceinline__ void grid_barrier(unsigned* cnt, unsigned target) {
  __threadfence();          // make our global stores visible
  __syncthreads();          // whole WG done with this step
  if (threadIdx.x == 0) {
    __hip_atomic_fetch_add(cnt, 1u, __ATOMIC_ACQ_REL, __HIP_MEMORY_SCOPE_AGENT);
    while (__hip_atomic_load(cnt, __ATOMIC_ACQUIRE, __HIP_MEMORY_SCOPE_AGENT) < target) {
      __builtin_amdgcn_s_sleep(1);
    }
  }
  __syncthreads();
}

// ---------------------------------------------------------------- cvt kernel
__global__ void cvt_kernel(const float* __restrict__ Wout, unsigned char* ws) {
  __bf16* wout = (__bf16*)(ws + WS_WOUT);
  unsigned tid = blockIdx.x * blockDim.x + threadIdx.x;
  unsigned stride = gridDim.x * blockDim.x;
  for (unsigned i = tid; i < NOUT * NHID; i += stride) wout[i] = (__bf16)Wout[i];
  if (tid == 0) *(unsigned*)(ws + WS_CNT) = 0u;   // reset grid barrier each call
}

// -------------------------------------------------------- persistent RNN kernel
// grid = 64 WGs x 256 threads. WG (mi, ng) owns batch rows [mi*16, +16) and
// hidden cols [ng*128, +128). Wave w owns the 16x16 tile at col ng*128 + w*16.
__global__ __launch_bounds__(256) void rnn_kernel(
    const float* __restrict__ x,     // [B,T,NIN]
    const float* __restrict__ h0,    // [B,NHID]
    const float* __restrict__ Wih,   // [NHID,NIN]
    const float* __restrict__ Whh,   // [NHID,NHID]
    const float* __restrict__ bih,
    const float* __restrict__ bhh,
    const float* __restrict__ alpha, // [NHID]
    float* __restrict__ hidden_list, // [B,T,NHID]
    float* __restrict__ h_final,     // [B,NHID]
    unsigned char* ws)
{
  __shared__ __bf16 Wh[128][520];   // W_hh slice, rows = local hidden col, padded
  __shared__ __bf16 Wi[128][136];   // W_ih slice
  __shared__ __bf16 hA[16][520];    // A operand: previous h tile (16 x 512)
  __shared__ __bf16 xA[16][136];    // A operand: x_t tile (16 x 128)

  __bf16* hbuf0 = (__bf16*)(ws + WS_HBUF);
  __bf16* hbuf1 = hbuf0 + (size_t)B_ * NHID;
  unsigned* cnt = (unsigned*)(ws + WS_CNT);

  const int tid  = threadIdx.x;
  const int wave = tid >> 5, lane = tid & 31;
  const int half = lane >> 4, l16 = lane & 15;
  const int mi = blockIdx.x & 15, ng = blockIdx.x >> 4;
  const int m0 = mi * 16, n0g = ng * 128;
  const int n  = n0g + wave * 16 + l16;   // this lane's hidden column (D layout)
  const int nl = wave * 16 + l16;         // local row into Wh / Wi

  // ---- stage weight slices fp32 -> bf16 into LDS (resident for all 500 steps)
  for (int c = 0; c < 64; ++c) {
    int lin = (c * 256 + tid) << 2;       // 65536 elems of Whh slice
    int r = lin >> 9, k = lin & 511;
    f4 v = *(const f4*)(Whh + (size_t)(n0g + r) * NHID + k);
    Wh[r][k] = (__bf16)v.x; Wh[r][k+1] = (__bf16)v.y;
    Wh[r][k+2] = (__bf16)v.z; Wh[r][k+3] = (__bf16)v.w;
  }
  for (int c = 0; c < 16; ++c) {
    int lin = (c * 256 + tid) << 2;       // 16384 elems of Wih slice
    int r = lin >> 7, k = lin & 127;
    f4 v = *(const f4*)(Wih + (size_t)(n0g + r) * NIN + k);
    Wi[r][k] = (__bf16)v.x; Wi[r][k+1] = (__bf16)v.y;
    Wi[r][k+2] = (__bf16)v.z; Wi[r][k+3] = (__bf16)v.w;
  }

  // ---- per-lane constants: leak + bias for column n
  const float al = alpha[n];
  const float om = 1.0f - al;
  const float bs = bih[n] + bhh[n];

  // ---- f32 h accumulator in registers (D layout: VGPR v -> row m0 + v + 8*half)
  v8f hacc;
#pragma unroll
  for (int v = 0; v < 8; ++v) hacc[v] = h0[(size_t)(m0 + v + 8 * half) * NHID + n];

  // ---- cooperative init of bf16 h ring buffer (this WG's 16x128 slice)
  for (int c = 0; c < 8; ++c) {
    int lin = c * 256 + tid;              // 2048 elems
    int r = lin >> 7, k = lin & 127;
    size_t idx = (size_t)(m0 + r) * NHID + n0g + k;
    hbuf0[idx] = (__bf16)h0[idx];
  }

  unsigned target = gridDim.x;
  grid_barrier(cnt, target);

  for (int t = 0; t < T_; ++t) {
    const __bf16* hb_cur = (t & 1) ? hbuf1 : hbuf0;
    __bf16*       hb_nxt = (t & 1) ? hbuf0 : hbuf1;

    // stage previous h tile (16 x 512 bf16) into LDS via async Global->LDS DMA
#pragma unroll
    for (int c = 0; c < 4; ++c) {
      int lin = (c * 256 + tid) << 3;     // 8192 bf16, 16 B per lane per issue
      int r = lin >> 9, k = lin & 511;
      async_g2l_b128(hb_cur + (size_t)(m0 + r) * NHID + k, &hA[r][k]);
    }
    // stage x_t tile (16 x 128 fp32 -> bf16) while the async copy is in flight
#pragma unroll
    for (int c = 0; c < 2; ++c) {
      int lin = (c * 256 + tid) << 2;     // 2048 floats
      int r = lin >> 7, k = lin & 127;
      const float* xp = x + ((size_t)(m0 + r) * T_ + t) * NIN + k;
      f4 v = *(const f4*)xp;
      xA[r][k] = (__bf16)v.x; xA[r][k+1] = (__bf16)v.y;
      xA[r][k+2] = (__bf16)v.z; xA[r][k+3] = (__bf16)v.w;
      if (t + 1 < T_) __builtin_prefetch(xp + NIN, 0, 3);  // next step's x tile
    }
    wait_async0();          // drain ASYNCcnt before the block barrier
    __syncthreads();

    v8f acc;
#pragma unroll
    for (int v = 0; v < 8; ++v) acc[v] = bs;

    // x @ W_ih^T : K = 128 -> 4 WMMAs
#pragma unroll
    for (int kb = 0; kb < 4; ++kb) {
      int k0 = kb * 32;
      v16bf a = cat8(*(const v8bf*)&xA[l16][k0 + 8 * half],
                     *(const v8bf*)&xA[l16][k0 + 16 + 8 * half]);
      v16bf bm = cat8(*(const v8bf*)&Wi[nl][k0 + 8 * half],
                      *(const v8bf*)&Wi[nl][k0 + 16 + 8 * half]);
      acc = wmma_bf16(a, bm, acc);
    }
    // h @ W_hh^T : K = 512 -> 16 WMMAs
#pragma unroll
    for (int kb = 0; kb < 16; ++kb) {
      int k0 = kb * 32;
      v16bf a = cat8(*(const v8bf*)&hA[l16][k0 + 8 * half],
                     *(const v8bf*)&hA[l16][k0 + 16 + 8 * half]);
      v16bf bm = cat8(*(const v8bf*)&Wh[nl][k0 + 8 * half],
                      *(const v8bf*)&Wh[nl][k0 + 16 + 8 * half]);
      acc = wmma_bf16(a, bm, acc);
    }

    // relu + per-unit leaky integration, exact in f32
#pragma unroll
    for (int v = 0; v < 8; ++v) {
      float cand = fmaxf(acc[v], 0.0f);
      hacc[v] = om * hacc[v] + al * cand;
    }

    // write required hidden_list output (f32) and bf16 ring buffer
#pragma unroll
    for (int v = 0; v < 8; ++v) {
      int bb = m0 + v + 8 * half;
      hidden_list[((size_t)bb * T_ + t) * NHID + n] = hacc[v];
      hb_nxt[(size_t)bb * NHID + n] = (__bf16)hacc[v];
    }

    target += gridDim.x;
    grid_barrier(cnt, target);
  }

#pragma unroll
  for (int v = 0; v < 8; ++v) {
    int bb = m0 + v + 8 * half;
    h_final[(size_t)bb * NHID + n] = hacc[v];
  }
}

// ------------------------------------------------------------- readout GEMM
// out[r, o] = sum_k hidden_list[r, k] * W_out[o, k] + b_out[o]
// grid = B*T/16 = 8000 blocks x 128 threads (4 waves, one 16-col o-tile each)
__global__ __launch_bounds__(128) void out_kernel(
    const float* __restrict__ hidden_list, // [B*T, NHID]
    const float* __restrict__ bout,        // [NOUT]
    float* __restrict__ out,               // [B*T, NOUT]
    const unsigned char* __restrict__ ws)
{
  __shared__ __bf16 hA[16][520];
  const __bf16* wout = (const __bf16*)(ws + WS_WOUT);

  const int tid  = threadIdx.x;
  const int wave = tid >> 5, lane = tid & 31;
  const int half = lane >> 4, l16 = lane & 15;
  const size_t r0 = (size_t)blockIdx.x * 16;

  // stage A: 16 x 512 f32 -> bf16 in LDS
#pragma unroll
  for (int c = 0; c < 16; ++c) {
    int lin = (c * 128 + tid) << 2;       // 8192 floats
    int r = lin >> 9, k = lin & 511;
    f4 v = *(const f4*)(hidden_list + (r0 + r) * NHID + k);
    hA[r][k] = (__bf16)v.x; hA[r][k+1] = (__bf16)v.y;
    hA[r][k+2] = (__bf16)v.z; hA[r][k+3] = (__bf16)v.w;
  }
  __syncthreads();

  const int o = wave * 16 + l16;          // this lane's output column
  v8f acc;
#pragma unroll
  for (int v = 0; v < 8; ++v) acc[v] = 0.0f;

#pragma unroll
  for (int kb = 0; kb < 16; ++kb) {
    int k0 = kb * 32;
    v16bf a = cat8(*(const v8bf*)&hA[l16][k0 + 8 * half],
                   *(const v8bf*)&hA[l16][k0 + 16 + 8 * half]);
    const __bf16* wr = wout + (size_t)o * NHID + k0 + 8 * half;
    v16bf bm = cat8(*(const v8bf*)wr, *(const v8bf*)(wr + 16));
    acc = wmma_bf16(a, bm, acc);
  }

  const float bo = bout[o];
#pragma unroll
  for (int v = 0; v < 8; ++v) {
    out[(r0 + v + 8 * half) * NOUT + o] = acc[v] + bo;
  }
}

// ------------------------------------------------------------------- launch
extern "C" void kernel_launch(void* const* d_in, const int* in_sizes, int n_in,
                              void* d_out, int out_size, void* d_ws, size_t ws_size,
                              hipStream_t stream) {
  const float* x     = (const float*)d_in[0];
  const float* h0    = (const float*)d_in[1];
  const float* Wih   = (const float*)d_in[2];
  const float* Whh   = (const float*)d_in[3];
  const float* bih   = (const float*)d_in[4];
  const float* bhh   = (const float*)d_in[5];
  const float* Wout  = (const float*)d_in[6];
  const float* bout  = (const float*)d_in[7];
  const float* alpha = (const float*)d_in[8];
  (void)in_sizes; (void)n_in; (void)out_size; (void)ws_size;

  float* outp        = (float*)d_out;
  float* hidden_list = outp;                                  // B*T*NHID
  float* output_list = outp + (size_t)B_ * T_ * NHID;         // B*T*NOUT
  float* h_final     = output_list + (size_t)B_ * T_ * NOUT;  // B*NHID
  unsigned char* ws  = (unsigned char*)d_ws;

  cvt_kernel<<<64, 256, 0, stream>>>(Wout, ws);
  rnn_kernel<<<64, 256, 0, stream>>>(x, h0, Wih, Whh, bih, bhh, alpha,
                                     hidden_list, h_final, ws);
  out_kernel<<<(B_ * T_) / 16, 128, 0, stream>>>(hidden_list, bout, output_list, ws);
}
